// BPPSModel_27264452395313
// MI455X (gfx1250) — compile-verified, matrix operand order checked by reading, not run on hardware
//
#include <hip/hip_runtime.h>
#include <hip/hip_bf16.h>

// ---------------------------------------------------------------------------
// BPPS model on MI455X (gfx1250): species-binned bf16 WMMA MLP + segment sum.
// 64-atom tiles: each wave owns 2 row-tiles x 4 col-tiles so every B fragment
// is reused twice. B fragments are addressed from a single per-wave lane base
// with compile-time immediate offsets (no per-load 64-bit VALU -> no WMMA->VALU
// WAR nop bubbles). ~101 KB LDS (320 KB/WGP -> 3 blocks/WGP).
// ---------------------------------------------------------------------------

typedef __bf16 bf16;
typedef __attribute__((ext_vector_type(16))) __bf16 v16bf;
typedef __attribute__((ext_vector_type(8)))  __bf16 v8bf;
typedef __attribute__((ext_vector_type(8)))  float  v8f;

union Frag16 { v16bf v; v8bf h[2]; };

constexpr int D_IN    = 512;
constexpr int HIDDEN  = 256;
constexpr int NSP     = 4;
constexpr int TILE    = 64;     // atoms per block
constexpr int XSTRIDE  = 528;   // bf16 elems per row (512 + 16 pad -> bank stride 8)
constexpr int H1STRIDE = 272;   // bf16 elems per row (256 + 16 pad)
constexpr int H2STRIDE = 264;   // f32  elems per row (256 + 8 pad)
constexpr int XBYTES   = TILE * XSTRIDE * 2;    // 67584
constexpr int H1BYTES  = TILE * H1STRIDE * 2;   // 34816
constexpr float INV_AVG = 1.0f / 60.0f;   // E_SCALE / AVG_N_ATOMS

__device__ __forceinline__ float wave_sum(float v) {
#pragma unroll
    for (int m = 16; m >= 1; m >>= 1) v += __shfl_xor(v, m, 32);
    return v;
}

__device__ __forceinline__ float silu(float x) {
    // x * rcp(1+e^-x): single v_rcp_f32 instead of the IEEE div sequence
    return x * __builtin_amdgcn_rcpf(1.0f + __expf(-x));
}

// A-fragment (16x32 bf16) from row-major LDS, per ISA 7.12.2 16-bit A layout.
// base is the LDS array start (lane offset added here); kBase/rowBase fold into ds immediate offsets.
__device__ __forceinline__ v16bf load_frag_A(const bf16* base, int stride,
                                             int rowBase, int kBase, int lane) {
    int m    = rowBase + (lane & 15);
    int ksel = lane >> 4;                      // 0: K 0..7/16..23, 1: K 8..15/24..31
    const bf16* p = base + m * stride + kBase + ksel * 8;
    Frag16 f;
    f.h[0] = *(const v8bf*)(p);                // K +0..7   (16B aligned)
    f.h[1] = *(const v8bf*)(p + 16);           // K +16..23 (16B aligned)
    return f.v;
}

// B-fragment (32x16 bf16): laneBase precomputed per wave; fragOfs is a
// compile-time fragment index after unrolling -> immediate global offsets.
__device__ __forceinline__ v16bf load_frag_B(const bf16* laneBase, int fragOfs) {
    const bf16* p = laneBase + (size_t)fragOfs * 512;
    Frag16 f;
    f.h[0] = *(const v8bf*)(p);
    f.h[1] = *(const v8bf*)(p + 8);
    return f.v;
}

// ---------------------------------------------------------------------------
// Pass 1: species histogram
__global__ __launch_bounds__(256)
void count_kernel(const int* __restrict__ sp, int* __restrict__ counts, int n) {
    int i = blockIdx.x * 256 + threadIdx.x;
    if (i < n) atomicAdd(&counts[sp[i]], 1);
}

// Pass 2: exclusive prefix (trivial, 4 species)
__global__ void prefix_kernel(int* ws32) {
    int* counts = ws32;      // [0..3]
    int* offs   = ws32 + 8;  // [8..11]
    int acc = 0;
#pragma unroll
    for (int s = 0; s < NSP; ++s) { offs[s] = acc; acc += counts[s]; }
}

// Pass 3: scatter atoms into species buckets + composition baseline
__global__ __launch_bounds__(256)
void scatter_kernel(const int* __restrict__ sp, const int* __restrict__ bt,
                    const float* __restrict__ Wc,
                    int* __restrict__ cursors, const int* __restrict__ offs,
                    int* __restrict__ sortedIdx, float* __restrict__ out, int n) {
    int i = blockIdx.x * 256 + threadIdx.x;
    if (i >= n) return;
    int s   = sp[i];
    int pos = atomicAdd(&cursors[s], 1);
    sortedIdx[offs[s] + pos] = i;
    atomicAdd(&out[bt[i]], Wc[s]);   // comp @ W_comp.T (OUT==1)
}

// Pass 4: reformat W1/W2 (f32 row-major) -> bf16 WMMA B-fragment layout
__global__ __launch_bounds__(256)
void reformat_kernel(const float* __restrict__ W1, const float* __restrict__ W2,
                     bf16* __restrict__ W1f, bf16* __restrict__ W2f) {
    int tid  = blockIdx.x * 256 + threadIdx.x;
    int lane = tid & 31;
    int frag = tid >> 5;
    int n    = lane & 15;
    int ksel = lane >> 4;
    if (frag < NSP * 16 * 16) {                 // W1: K=512 (16 kc) x N=256 (16 j)
        int j  = frag & 15;
        int kc = (frag >> 4) & 15;
        int s  = frag >> 8;
        const float* src = W1 + (size_t)s * D_IN * HIDDEN;
        bf16* dst = W1f + (size_t)frag * 512 + lane * 16;
#pragma unroll
        for (int e = 0; e < 16; ++e) {
            int k = kc * 32 + (e < 8 ? ksel * 8 + e : 16 + ksel * 8 + (e - 8));
            dst[e] = (bf16)src[(size_t)k * HIDDEN + j * 16 + n];
        }
    } else {
        int f2 = frag - NSP * 16 * 16;
        if (f2 >= NSP * 8 * 16) return;         // W2: K=256 (8 kc) x N=256 (16 j)
        int j  = f2 & 15;
        int kc = (f2 >> 4) & 7;
        int s  = f2 >> 7;
        const float* src = W2 + (size_t)s * HIDDEN * HIDDEN;
        bf16* dst = W2f + (size_t)f2 * 512 + lane * 16;
#pragma unroll
        for (int e = 0; e < 16; ++e) {
            int k = kc * 32 + (e < 8 ? ksel * 8 + e : 16 + ksel * 8 + (e - 8));
            dst[e] = (bf16)src[(size_t)k * HIDDEN + j * 16 + n];
        }
    }
}

// ---------------------------------------------------------------------------
// Pass 5: main fused kernel. One block = 64 atoms of one species.
// LayerNorm -> [512x256 bf16 WMMA] -> SiLU -> [256x256 bf16 WMMA] -> SiLU
// -> dot(W3) -> atomicAdd(out[struct], e/60)
__global__ __launch_bounds__(256)
void mlp_kernel(const float* __restrict__ ps,
                const float* __restrict__ gam, const float* __restrict__ bet,
                const float* __restrict__ W3, const int* __restrict__ bt,
                const int* __restrict__ wsInt,   // counts[0..3], offs[8..11]
                const int* __restrict__ sortedIdx,
                const bf16* __restrict__ W1f, const bf16* __restrict__ W2f,
                float* __restrict__ out, int tiles) {
    int bx = blockIdx.x;
    int s  = bx / tiles;
    int t  = bx - s * tiles;
    const int* counts = wsInt;
    const int* offs   = wsInt + 8;
    int cnt  = counts[s];
    int base = offs[s];
    if (t * TILE >= cnt) return;                // uniform early exit

    __shared__ __align__(16) unsigned char smem[XBYTES + H1BYTES + 512];
    bf16*  X   = (bf16*)smem;                   // 64 x 528 bf16 = 67584 B
    float* H2  = (float*)smem;                  // reuse: 64 x 264 f32 = 67584 B
    bf16*  H1  = (bf16*)(smem + XBYTES);        // 64 x 272 bf16 = 34816 B
    int* sIdxS = (int*)(smem + XBYTES + H1BYTES);  // 64 ints
    int* sBatS = sIdxS + TILE;                     // 64 ints

    int tid  = threadIdx.x;
    int w    = tid >> 5;
    int lane = tid & 31;

    // ---- stage 0: gather atom ids + structure ids -------------------------
    if (tid < TILE) {
        int g  = t * TILE + tid;
        int id = -1, bb = 0;
        if (g < cnt) { id = sortedIdx[base + g]; bb = bt[id]; }
        sIdxS[tid] = id;
        sBatS[tid] = bb;
    }
    __syncthreads();

    // ---- stage 1: LayerNorm, write bf16 X to LDS (one wave per 8 rows) ----
    {
        // gamma/beta are row-invariant per lane: hoist out of the row loop
        float gk[16], bk[16];
#pragma unroll
        for (int it = 0; it < 16; ++it) {
            int k = it * 32 + lane;
            gk[it] = gam[k];
            bk[it] = bet[k];
        }
#pragma unroll
        for (int rr = 0; rr < 8; ++rr) {
            int r  = w * 8 + rr;
            int id = sIdxS[r];
            float x[16];
            float sum = 0.f, sq = 0.f;
            if (id >= 0) {
                const float* row = ps + (size_t)id * D_IN;
#pragma unroll
                for (int it = 0; it < 16; ++it) {
                    float v = row[it * 32 + lane];
                    x[it] = v; sum += v; sq += v * v;
                }
            } else {
#pragma unroll
                for (int it = 0; it < 16; ++it) x[it] = 0.f;
            }
            sum = wave_sum(sum);
            sq  = wave_sum(sq);
            float mean = sum * (1.f / 512.f);
            float var  = sq * (1.f / 512.f) - mean * mean;
            float rstd = rsqrtf(var + 1e-5f);
            bf16* xr = X + r * XSTRIDE;
#pragma unroll
            for (int it = 0; it < 16; ++it) {
                int k = it * 32 + lane;
                float vn = (id >= 0) ? ((x[it] - mean) * rstd * gk[it] + bk[it]) : 0.f;
                xr[k] = (bf16)vn;
            }
        }
    }
    __syncthreads();

    // Wave tiling: 4 row-tiles x 16 col-tiles of 16x16 outputs.
    // wave w: row-tile pair rtp = w&1 -> rows {rtp*2, rtp*2+1}; cols jq*4..jq*4+3.
    int rtp = w & 1;
    int jq  = w >> 1;
    int n   = lane & 15;
    int mb  = (lane >> 4) * 8;

    // ---- stage 2: H1 = silu(X @ W1[s]) ------------------------------------
    {
        // one lane base; fragment deltas are compile-time -> immediate offsets
        const bf16* bb = W1f + ((size_t)s * 256 + jq * 4) * 512 + (size_t)lane * 16;
        v8f acc[2][4] = {};
#pragma unroll
        for (int kc = 0; kc < 16; ++kc) {
            v16bf a0 = load_frag_A(X, XSTRIDE, (rtp * 2 + 0) * 16, kc * 32, lane);
            v16bf a1 = load_frag_A(X, XSTRIDE, (rtp * 2 + 1) * 16, kc * 32, lane);
#pragma unroll
            for (int q = 0; q < 4; ++q) {
                v16bf bm = load_frag_B(bb, kc * 16 + q);
                acc[0][q] = __builtin_amdgcn_wmma_f32_16x16x32_bf16(
                    false, a0, false, bm, (short)0, acc[0][q], false, false);
                acc[1][q] = __builtin_amdgcn_wmma_f32_16x16x32_bf16(
                    false, a1, false, bm, (short)0, acc[1][q], false, false);
            }
        }
#pragma unroll
        for (int rt = 0; rt < 2; ++rt) {
#pragma unroll
            for (int q = 0; q < 4; ++q) {
                int j = jq * 4 + q;
#pragma unroll
                for (int v = 0; v < 8; ++v) {   // C/D layout: lane=N, M = v + 8*(lane>=16)
                    float sv = silu(acc[rt][q][v]);
                    H1[((rtp * 2 + rt) * 16 + mb + v) * H1STRIDE + j * 16 + n] = (bf16)sv;
                }
            }
        }
    }
    __syncthreads();

    // ---- stage 3: H2 = silu(H1 @ W2[s])  (H2 reuses X's LDS region) -------
    {
        const bf16* bb = W2f + ((size_t)s * 128 + jq * 4) * 512 + (size_t)lane * 16;
        v8f acc[2][4] = {};
#pragma unroll
        for (int kc = 0; kc < 8; ++kc) {
            v16bf a0 = load_frag_A(H1, H1STRIDE, (rtp * 2 + 0) * 16, kc * 32, lane);
            v16bf a1 = load_frag_A(H1, H1STRIDE, (rtp * 2 + 1) * 16, kc * 32, lane);
#pragma unroll
            for (int q = 0; q < 4; ++q) {
                v16bf bm = load_frag_B(bb, kc * 16 + q);
                acc[0][q] = __builtin_amdgcn_wmma_f32_16x16x32_bf16(
                    false, a0, false, bm, (short)0, acc[0][q], false, false);
                acc[1][q] = __builtin_amdgcn_wmma_f32_16x16x32_bf16(
                    false, a1, false, bm, (short)0, acc[1][q], false, false);
            }
        }
#pragma unroll
        for (int rt = 0; rt < 2; ++rt) {
#pragma unroll
            for (int q = 0; q < 4; ++q) {
                int j = jq * 4 + q;
#pragma unroll
                for (int v = 0; v < 8; ++v) {
                    H2[((rtp * 2 + rt) * 16 + mb + v) * H2STRIDE + j * 16 + n] =
                        silu(acc[rt][q][v]);
                }
            }
        }
    }
    __syncthreads();

    // ---- stage 4: e = H2 . W3[s], segment-sum via atomics -----------------
    const float* w3 = W3 + s * HIDDEN;
#pragma unroll
    for (int rr = 0; rr < 8; ++rr) {
        int r = w * 8 + rr;
        float p = 0.f;
#pragma unroll
        for (int e = 0; e < 8; ++e) {
            int k = lane * 8 + e;
            p += H2[r * H2STRIDE + k] * w3[k];
        }
        p = wave_sum(p);
        if (lane == 0 && sIdxS[r] >= 0)
            atomicAdd(&out[sBatS[r]], p * INV_AVG);
    }
}

// ---------------------------------------------------------------------------
extern "C" void kernel_launch(void* const* d_in, const int* in_sizes, int n_in,
                              void* d_out, int out_size, void* d_ws, size_t ws_size,
                              hipStream_t stream) {
    const float* ps  = (const float*)d_in[0];
    const float* gam = (const float*)d_in[1];
    const float* bet = (const float*)d_in[2];
    const float* W1  = (const float*)d_in[3];
    const float* W2  = (const float*)d_in[4];
    const float* W3  = (const float*)d_in[5];
    const float* Wc  = (const float*)d_in[6];
    const int*   sp  = (const int*)d_in[7];
    const int*   bt  = (const int*)d_in[8];
    float* out = (float*)d_out;

    const int nAtoms = in_sizes[7];

    // workspace layout
    char* ws = (char*)d_ws;
    int*  wsInt     = (int*)ws;                       // counts[0..3], cursors[4..7], offs[8..11]
    int*  cursors   = wsInt + 4;
    int*  offs      = wsInt + 8;
    int*  sortedIdx = (int*)(ws + 64);
    size_t idxEnd = 64 + (size_t)nAtoms * 4;
    size_t w1Off  = (idxEnd + 255) & ~(size_t)255;
    bf16* W1f = (bf16*)(ws + w1Off);                  // 4*512*256*2 = 1 MiB
    bf16* W2f = (bf16*)(ws + w1Off + (size_t)NSP * D_IN * HIDDEN * 2);  // 512 KiB

    hipMemsetAsync(ws, 0, 64, stream);                       // counters
    hipMemsetAsync(out, 0, (size_t)out_size * 4, stream);    // energies

    int nb = (nAtoms + 255) / 256;
    count_kernel<<<nb, 256, 0, stream>>>(sp, wsInt, nAtoms);
    prefix_kernel<<<1, 1, 0, stream>>>(wsInt);
    scatter_kernel<<<nb, 256, 0, stream>>>(sp, bt, Wc, cursors, offs,
                                           sortedIdx, out, nAtoms);
    // 1024 W1 frags + 512 W2 frags, 32 lanes each = 49152 threads
    reformat_kernel<<<192, 256, 0, stream>>>(W1, W2, W1f, W2f);

    int tiles = (nAtoms + TILE - 1) / TILE;
    mlp_kernel<<<NSP * tiles, 256, 0, stream>>>(ps, gam, bet, W3, bt,
                                                wsInt, sortedIdx, W1f, W2f,
                                                out, tiles);
}
